// TransformerFeedForward_84825604096023
// MI455X (gfx1250) — matrix operand order repaired and checked by reading, not compiled, not run on hardware
//
#include <hip/hip_runtime.h>

typedef __attribute__((ext_vector_type(16))) _Float16 v16h;
typedef __attribute__((ext_vector_type(8)))  float    v8f;

#define T_STEPS 4
#define BN      16384      // B*N = 64*256
#define DDIM    512
#define FDIM    2048
#define KSTEP   32

// -------------------------------------------------------------------------
// Workspace layout (84 MB total):
//   [0,     2MB) : W_fc   as f16  (2048 x 512, row-major)
//   [2MB,   4MB) : W_proj as f16  (512 x 2048, row-major)
//   [4MB,  20MB) : spike bitmasks u32 [T*BN][F/32] (bit i of word w <-> f=32w+i)
//   [20MB, 84MB) : x as f16 [T, BN, D]
// -------------------------------------------------------------------------

__global__ __launch_bounds__(256) void snn_cvt_f32_f16(
    const float* __restrict__ s, _Float16* __restrict__ d, int n)
{
    int i = (blockIdx.x * 256 + threadIdx.x) * 4;   // n is a multiple of 4
    if (i < n) {
        float4 v = *(const float4*)(s + i);
        d[i + 0] = (_Float16)v.x;
        d[i + 1] = (_Float16)v.y;
        d[i + 2] = (_Float16)v.z;
        d[i + 3] = (_Float16)v.w;
    }
}

// -------------------------------------------------------------------------
// Kernel 1: h = x @ W_fc^T  (f16 WMMA, fp32 acc), LIF over T in registers,
// spikes bit-packed via __ballot. x tiles staged into double-buffered LDS
// with global_load_async_to_lds_b128 (ASYNCcnt), branchless software
// pipeline: always issue next buffer (k wraps), wait <=1, barrier, compute.
// Block: 16 (b,n)-rows x 256 F-cols. 8 waves, each wave: 2 F-tiles x 4 t.
// -------------------------------------------------------------------------
__global__ __launch_bounds__(256) void snn_ffn_gemm1_lif(
    const _Float16* __restrict__ xH,        // [T, BN, D] f16
    const _Float16* __restrict__ WfcH,      // [F, D] f16
    unsigned* __restrict__ sbits32)         // [T*BN][F/32] bitmasks
{
    __shared__ __align__(64) _Float16 ldsX[2][T_STEPS][16][KSTEP];  // 2 x 4 KB
    const int tid  = threadIdx.x;
    const int m0   = blockIdx.x * 16;
    const int f0   = blockIdx.y * 256;
    const int wave = tid >> 5;
    const int lane = tid & 31;
    const int fw   = f0 + wave * 32;        // wave's 2 F-tiles
    const int mrow = lane & 15;
    const int koff   = (lane >> 4) * 8;     // A: contiguous K-half [khalfB..+15] per lane-half
    const int khalfB = (lane >> 4) * 16;    // B: contiguous K-half per lane-half

    // Per-thread async-copy source: 16 bytes (8 f16) of one x row.
    // Linear element e0 = tid*8 over [t][m][kk] with kk in {0,8,16,24}.
    const int e0 = tid << 3;
    const int ckk = e0 & 31, cm = (e0 >> 5) & 15, ct = e0 >> 9;
    const _Float16* srcBase =
        xH + (((size_t)ct * BN + (m0 + cm)) * DDIM + ckk);
    const unsigned ldsBase =
        (unsigned)(uintptr_t)(&ldsX[0][0][0][0]) + (unsigned)(tid << 4);

    v8f c[2][T_STEPS];
    #pragma unroll
    for (int ft = 0; ft < 2; ++ft)
        for (int t = 0; t < T_STEPS; ++t)
            for (int i = 0; i < 8; ++i) c[ft][t][i] = 0.0f;

    // Prologue: async-stage k=0 tile into buffer 0
    {
        unsigned long long ga = (unsigned long long)(uintptr_t)srcBase;
        asm volatile("global_load_async_to_lds_b128 %0, %1, off"
                     :: "v"(ldsBase), "v"(ga) : "memory");
    }

    int buf = 0;
    #pragma unroll 2
    for (int k = 0; k < DDIM; k += KSTEP) {
        // Branchless pipeline: always issue next tile into the other buffer.
        // Last iteration wraps to k=0 (dead copy, never read; endpgm waits idle).
        int kn = (k + KSTEP) & (DDIM - 1);
        {
            unsigned long long ga =
                (unsigned long long)(uintptr_t)(srcBase + kn);
            unsigned ldst = ldsBase + (unsigned)((buf ^ 1) * 4096);
            asm volatile("global_load_async_to_lds_b128 %0, %1, off"
                         :: "v"(ldst), "v"(ga) : "memory");
        }
        // In-order async completion: <=1 outstanding => current buffer ready.
        asm volatile("s_wait_asynccnt 0x1" ::: "memory");
        __syncthreads();    // all waves' copies of `buf` complete

        // A fragments per timestep (contiguous order: element s of lane-half h <-> k = 16h + s)
        v16h a[T_STEPS];
        #pragma unroll
        for (int t = 0; t < T_STEPS; ++t) {
            const _Float16* ap = &ldsX[buf][t][mrow][0];
            #pragma unroll
            for (int i = 0; i < 8; ++i) {
                a[t][i]     = ap[khalfB + i];
                a[t][8 + i] = ap[khalfB + 8 + i];
            }
        }
        #pragma unroll
        for (int ft = 0; ft < 2; ++ft) {
            // B fragment: lane n holds W_fc[f, k+khalfB .. +15] (contiguous row)
            v16h b = *(const v16h*)(WfcH + (size_t)(fw + ft * 16 + mrow) * DDIM + k + khalfB);
            #pragma unroll
            for (int t = 0; t < T_STEPS; ++t)
                c[ft][t] = __builtin_amdgcn_wmma_f32_16x16x32_f16(
                    false, a[t], false, b, (short)0, c[ft][t], false, false);
        }
        __syncthreads();    // all waves done reading `buf` before it is refilled
        buf ^= 1;
    }

    // LIF: v <- v*(1 - 1/TAU) + h_t ; spike = (v >= 1) ; hard reset.
    // C layout: VGPR j is row m0+j (lanes 0-15) / m0+j+8 (lanes 16-31), lane%16 = col.
    // Ballot bit = lane = F column; select halves into lane-indexed regs, then
    // one u32 store per timestep (both F-tiles packed) with lanes 0..15 active.
    const float kDecay = 1.0f / 6.0f;       // 1 - 1/1.2
    unsigned sv[T_STEPS] = {0u, 0u, 0u, 0u};
    #pragma unroll
    for (int ft = 0; ft < 2; ++ft) {
        #pragma unroll
        for (int j = 0; j < 8; ++j) {
            float v = 0.0f;
            #pragma unroll
            for (int t = 0; t < T_STEPS; ++t) {
                v = v * kDecay + c[ft][t][j];
                bool spike = (v >= 1.0f);
                if (spike) v = 0.0f;
                unsigned mask = (unsigned)__ballot(spike);
                unsigned half = (lane & 8) ? (mask >> 16) : (mask & 0xFFFFu);
                if ((lane & 7) == j) sv[t] |= half << (16 * ft);
            }
        }
    }
    if (lane < 16) {
        int row = m0 + (lane & 7) + (lane & 8);
        #pragma unroll
        for (int t = 0; t < T_STEPS; ++t)
            sbits32[((size_t)t * BN + row) * (FDIM / 32) + (fw >> 5)] = sv[t];
    }
}

// -------------------------------------------------------------------------
// Kernel 2: out = spikes @ W_proj^T (A rebuilt from bitmasks), fused LayerNorm.
// Block: 16 rows x full D=512 cols (8 waves x 64 cols). LN via LDS + shfl.
// -------------------------------------------------------------------------
__global__ __launch_bounds__(256) void snn_ffn_gemm2_ln(
    const unsigned* __restrict__ sbits32,   // [T*BN][F/32]
    const _Float16* __restrict__ WprojH,    // [D, F] f16
    const float* __restrict__ gamma, const float* __restrict__ beta,
    float* __restrict__ out)                // [T*BN, D] fp32
{
    __shared__ __align__(64) float ldsOut[16][DDIM];   // 32 KB
    const int m0   = blockIdx.x * 16;
    const int wave = threadIdx.x >> 5;
    const int lane = threadIdx.x & 31;
    const int n0   = wave * 64;
    const int mrow = lane & 15;
    const int koff   = (lane >> 4) * 8;
    const int khalfB = (lane >> 4) * 16;

    v8f c[4];
    #pragma unroll
    for (int nt = 0; nt < 4; ++nt)
        for (int i = 0; i < 8; ++i) c[nt][i] = 0.0f;

    #pragma unroll 2
    for (int k = 0; k < FDIM; k += KSTEP) {
        // A fragment straight from spike bits: word w covers f = k .. k+31
        unsigned w = sbits32[(size_t)(m0 + mrow) * (FDIM / 32) + (k >> 5)];
        v16h a;
        #pragma unroll
        for (int i = 0; i < 8; ++i) {
            a[i]     = (_Float16)(float)((w >> (khalfB + i)) & 1u);
            a[8 + i] = (_Float16)(float)((w >> (khalfB + 8 + i)) & 1u);
        }
        #pragma unroll
        for (int nt = 0; nt < 4; ++nt) {
            v16h b = *(const v16h*)(WprojH + (size_t)(n0 + nt * 16 + mrow) * FDIM + k + khalfB);
            c[nt] = __builtin_amdgcn_wmma_f32_16x16x32_f16(
                false, a, false, b, (short)0, c[nt], false, false);
        }
    }

    // Spill tile to LDS in row-major for LayerNorm
    #pragma unroll
    for (int nt = 0; nt < 4; ++nt)
        #pragma unroll
        for (int j = 0; j < 8; ++j)
            ldsOut[j + ((lane >> 4) << 3)][n0 + nt * 16 + mrow] = c[nt][j];
    __syncthreads();

    // LayerNorm: 16 threads per row, interleaved columns (col = cg + 16*i)
    // -> conflict-free LDS reads, 64B-coalesced global stores.
    int row = threadIdx.x >> 4;
    int cg  = threadIdx.x & 15;
    float sum = 0.0f, sq = 0.0f;
    #pragma unroll 8
    for (int i = 0; i < 32; ++i) {
        float v = ldsOut[row][cg + 16 * i];
        sum += v; sq += v * v;
    }
    #pragma unroll
    for (int off = 8; off > 0; off >>= 1) {
        sum += __shfl_xor(sum, off, 16);
        sq  += __shfl_xor(sq,  off, 16);
    }
    float mu  = sum * (1.0f / DDIM);
    float var = sq  * (1.0f / DDIM) - mu * mu;
    float inv = rsqrtf(var + 1e-5f);
    size_t ob = (size_t)(m0 + row) * DDIM;
    #pragma unroll 8
    for (int i = 0; i < 32; ++i) {
        int col = cg + 16 * i;
        out[ob + col] = (ldsOut[row][col] - mu) * inv * gamma[col] + beta[col];
    }
}

// -------------------------------------------------------------------------
extern "C" void kernel_launch(void* const* d_in, const int* in_sizes, int n_in,
                              void* d_out, int out_size, void* d_ws, size_t ws_size,
                              hipStream_t stream) {
    (void)in_sizes; (void)n_in; (void)out_size; (void)ws_size;
    const float* x     = (const float*)d_in[0];   // [4,64,256,512]
    const float* Wfc   = (const float*)d_in[1];   // [2048,512]
    const float* Wproj = (const float*)d_in[2];   // [512,2048]
    const float* gamma = (const float*)d_in[3];   // [512]
    const float* beta  = (const float*)d_in[4];   // [512]
    float* out = (float*)d_out;

    char* ws = (char*)d_ws;
    _Float16* WfcH   = (_Float16*)(ws);
    _Float16* WprojH = (_Float16*)(ws + (size_t)2 * 1024 * 1024);
    unsigned* sbits  = (unsigned*)(ws + (size_t)4 * 1024 * 1024);
    _Float16* xH     = (_Float16*)(ws + (size_t)20 * 1024 * 1024);

    const int nW = FDIM * DDIM;               // 1,048,576
    const int nX = T_STEPS * BN * DDIM;       // 33,554,432
    snn_cvt_f32_f16<<<nW / (256 * 4), 256, 0, stream>>>(Wfc, WfcH, nW);
    snn_cvt_f32_f16<<<nW / (256 * 4), 256, 0, stream>>>(Wproj, WprojH, nW);
    snn_cvt_f32_f16<<<nX / (256 * 4), 256, 0, stream>>>(x, xH, nX);

    dim3 g1(BN / 16, FDIM / 256);             // (1024, 8)
    snn_ffn_gemm1_lif<<<g1, 256, 0, stream>>>(xH, WfcH, sbits);

    snn_ffn_gemm2_ln<<<(T_STEPS * BN) / 16, 256, 0, stream>>>(
        sbits, WprojH, gamma, beta, out);
}
